// QuantizedLinear_11776800325931
// MI455X (gfx1250) — 8-bit matrix operand order repaired and checked by reading, compiled, not run on hardware
//
#include <hip/hip_runtime.h>
#include <hip/hip_bf16.h>
#include <stdint.h>

typedef int   v4i  __attribute__((ext_vector_type(4)));
typedef int   v16i __attribute__((ext_vector_type(16)));
typedef float v8f  __attribute__((ext_vector_type(8)));

#define TOKENS   8192
#define DIN      4096
#define DOUT     4096
#define GROUP    128
#define NGROUPS  32
#define BM       64
#define BN       128

__device__ __forceinline__ unsigned int fp8_of_mag(int q) {
  const unsigned long long LO = 0x4E4C4A4844403800ULL;
  const unsigned long long HI = 0x5756555453525150ULL;
  unsigned long long t = (q < 8) ? (LO >> (8 * q)) : (HI >> (8 * (q - 8)));
  return (unsigned int)(t & 0xFFu);
}

__global__ __launch_bounds__(256)
void quant_act_kernel(const float* __restrict__ x,
                      unsigned char* __restrict__ xq8,
                      float* __restrict__ ascale,
                      float* __restrict__ mcorr) {
  __shared__ float smin[256];
  __shared__ float smax[256];
  const int t   = blockIdx.x;
  const int tid = threadIdx.x;
  const float4* __restrict__ xr = (const float4*)(x + (size_t)t * DIN);

  float4 v[4];
  float mn = 3.0e38f, mx = -3.0e38f;
#pragma unroll
  for (int i = 0; i < 4; ++i) {
    v[i] = xr[tid + i * 256];
    mn = fminf(mn, fminf(fminf(v[i].x, v[i].y), fminf(v[i].z, v[i].w)));
    mx = fmaxf(mx, fmaxf(fmaxf(v[i].x, v[i].y), fmaxf(v[i].z, v[i].w)));
  }
  smin[tid] = mn;
  smax[tid] = mx;
  __syncthreads();
  for (int s = 128; s > 0; s >>= 1) {
    if (tid < s) {
      smin[tid] = fminf(smin[tid], smin[tid + s]);
      smax[tid] = fmaxf(smax[tid], smax[tid + s]);
    }
    __syncthreads();
  }
  const float xmin = smin[0];
  const float xmax = smax[0];
  const float as = fmaxf((xmax - xmin) * (1.0f / 15.0f), 1e-8f);
  const float zp = rintf(-xmin / as);
  if (tid == 0) {
    ascale[t] = as;
    mcorr[t]  = as * zp;
  }

  unsigned int* __restrict__ dst = (unsigned int*)(xq8 + (size_t)t * DIN);
#pragma unroll
  for (int i = 0; i < 4; ++i) {
    const float f[4] = {v[i].x, v[i].y, v[i].z, v[i].w};
    unsigned int packed = 0;
#pragma unroll
    for (int b = 0; b < 4; ++b) {
      const float qf = rintf(f[b] / as + zp);
      const int q = (int)fminf(fmaxf(qf, 0.0f), 15.0f);
      packed |= fp8_of_mag(q) << (8 * b);
    }
    dst[tid + i * 256] = packed;
  }
}

__global__ __launch_bounds__(256)
void weight_prep_kernel(const signed char* __restrict__ qw,
                        const float* __restrict__ wscale,
                        unsigned char* __restrict__ wq8,
                        float* __restrict__ wsum) {
  const int lane = threadIdx.x & 31;
  const int wave = threadIdx.x >> 5;
  const int o = blockIdx.x * 8 + wave;

  const int* __restrict__ src = (const int*)(qw + (size_t)o * DIN + lane * GROUP);
  unsigned int* __restrict__ dst = (unsigned int*)(wq8 + (size_t)o * DIN + lane * GROUP);

  int s = 0;
#pragma unroll
  for (int c = 0; c < GROUP / 4; ++c) {
    const int w = src[c];
    unsigned int packed = 0;
#pragma unroll
    for (int b = 0; b < 4; ++b) {
      const int q = (int)(signed char)(w >> (8 * b));
      s += q;
      const unsigned int f8 = (q >= 0) ? fp8_of_mag(q) : (0x80u | fp8_of_mag(-q));
      packed |= f8 << (8 * b);
    }
    dst[c] = packed;
  }
  float partial = wscale[o * NGROUPS + lane] * (float)s;
#pragma unroll
  for (int m = 16; m >= 1; m >>= 1) partial += __shfl_xor(partial, m, 32);
  if (lane == 0) wsum[o] = partial;
}

union Frag {
  v16i v;
  v4i  q[4];
  unsigned long long d[8];
};

__global__ __launch_bounds__(256)
void gemm_fp8_kernel(const unsigned char* __restrict__ Xq,
                     const unsigned char* __restrict__ Wq,
                     const float* __restrict__ wscale,
                     const float* __restrict__ ascale,
                     const float* __restrict__ mcorr,
                     const float* __restrict__ wsum,
                     const float* __restrict__ bias,
                     float* __restrict__ out) {
  __shared__ unsigned char As[2][BM * GROUP];
  __shared__ unsigned char Bs[2][BN * GROUP];

  const int tid  = threadIdx.x;
  const int lane = tid & 31;
  const int wave = tid >> 5;
  const int wm   = wave & 1;
  const int wn   = wave >> 1;
  const int half = lane >> 4;
  const int l16  = lane & 15;
  const int m0 = blockIdx.x * BM;
  const int n0 = blockIdx.y * BN;

  v4i areg[2], breg[4];

  auto gload = [&](int g) {
    const size_t k0 = (size_t)g * GROUP;
#pragma unroll
    for (int i = 0; i < 2; ++i) {
      const int id = tid + i * 256;
      areg[i] = *(const v4i*)(Xq + (size_t)(m0 + (id >> 3)) * DIN + k0 + (id & 7) * 16);
    }
#pragma unroll
    for (int i = 0; i < 4; ++i) {
      const int id = tid + i * 256;
      breg[i] = *(const v4i*)(Wq + (size_t)(n0 + (id >> 3)) * DIN + k0 + (id & 7) * 16);
    }
  };
  auto lstore = [&](int buf) {
#pragma unroll
    for (int i = 0; i < 2; ++i) {
      const int id = tid + i * 256;
      *(v4i*)(&As[buf][(id >> 3) * GROUP + (id & 7) * 16]) = areg[i];
    }
#pragma unroll
    for (int i = 0; i < 4; ++i) {
      const int id = tid + i * 256;
      *(v4i*)(&Bs[buf][(id >> 3) * GROUP + (id & 7) * 16]) = breg[i];
    }
  };

  const v8f zero8 = {0.f, 0.f, 0.f, 0.f, 0.f, 0.f, 0.f, 0.f};
  v8f facc[2][2];
#pragma unroll
  for (int mt = 0; mt < 2; ++mt)
#pragma unroll
    for (int nt = 0; nt < 2; ++nt) facc[mt][nt] = zero8;

  gload(0);
  lstore(0);
  __syncthreads();

  for (int g = 0; g < NGROUPS; ++g) {
    const int buf = g & 1;
    if (g + 1 < NGROUPS) gload(g + 1);

    Frag af[2], bf[2];
#pragma unroll
    for (int mt = 0; mt < 2; ++mt) {
      const unsigned char* ap =
          &As[buf][(wm * 32 + mt * 16 + l16) * GROUP + half * 16];
#pragma unroll
      for (int j = 0; j < 8; ++j)
        af[mt].d[j] = *(const unsigned long long*)(ap + (j >> 1) * 32 + (j & 1) * 8);
    }
#pragma unroll
    for (int nt = 0; nt < 2; ++nt) {
      const unsigned char* bp =
          &Bs[buf][(wn * 32 + nt * 16 + l16) * GROUP + half * 16];
#pragma unroll
      for (int j = 0; j < 4; ++j)
        bf[nt].q[j] = *(const v4i*)(bp + j * 32);
    }

#pragma unroll
    for (int nt = 0; nt < 2; ++nt) {
      const float ws = wscale[(size_t)(n0 + wn * 32 + nt * 16 + l16) * NGROUPS + g];
#pragma unroll
      for (int mt = 0; mt < 2; ++mt) {
        const v8f d = __builtin_amdgcn_wmma_f32_16x16x128_fp8_fp8(
            af[mt].v, bf[nt].v, (short)0, zero8, false, false);
#pragma unroll
        for (int r = 0; r < 8; ++r) facc[mt][nt][r] += d[r] * ws;
      }
    }

    if (g + 1 < NGROUPS) lstore(buf ^ 1);
    __syncthreads();
  }

#pragma unroll
  for (int nt = 0; nt < 2; ++nt) {
    const int col = n0 + wn * 32 + nt * 16 + l16;
    const float bz  = bias[col];
    const float wsv = wsum[col];
#pragma unroll
    for (int mt = 0; mt < 2; ++mt) {
      const int rbase = m0 + wm * 32 + mt * 16 + half * 8;
#pragma unroll
      for (int r = 0; r < 8; ++r) {
        const int row = rbase + r;
        out[(size_t)row * DOUT + col] =
            ascale[row] * facc[mt][nt][r] - mcorr[row] * wsv + bz;
      }
    }
  }
}

extern "C" void kernel_launch(void* const* d_in, const int* in_sizes, int n_in,
                              void* d_out, int out_size, void* d_ws, size_t ws_size,
                              hipStream_t stream) {
  (void)in_sizes; (void)n_in; (void)out_size; (void)ws_size;
  const float*       x      = (const float*)d_in[0];
  const signed char* qw     = (const signed char*)d_in[1];
  const float*       wscale = (const float*)d_in[2];
  const float*       bias   = (const float*)d_in[3];
  float* out = (float*)d_out;

  unsigned char* ws8 = (unsigned char*)d_ws;
  unsigned char* Wq8 = ws8;
  unsigned char* Xq8 = ws8 + (size_t)DOUT * DIN;
  float* Ascale = (float*)(ws8 + (size_t)DOUT * DIN + (size_t)TOKENS * DIN);
  float* Mcorr  = Ascale + TOKENS;
  float* Wsum   = Mcorr + TOKENS;

  hipLaunchKernelGGL(quant_act_kernel, dim3(TOKENS), dim3(256), 0, stream,
                     x, Xq8, Ascale, Mcorr);
  hipLaunchKernelGGL(weight_prep_kernel, dim3(DOUT / 8), dim3(256), 0, stream,
                     qw, wscale, Wq8, Wsum);
  hipLaunchKernelGGL(gemm_fp8_kernel, dim3(TOKENS / BM, DOUT / BN), dim3(256), 0, stream,
                     Xq8, Wq8, wscale, Ascale, Mcorr, Wsum, bias, out);
}
